// EnergySNN_34909494181995
// MI455X (gfx1250) — matrix operand order repaired and checked by reading, not compiled, not run on hardware
//
#include <hip/hip_runtime.h>

// ---------------------------------------------------------------------------
// Types for CDNA5 WMMA (wave32, 16x16x32 bf16 -> f32)
// ---------------------------------------------------------------------------
typedef __attribute__((ext_vector_type(8)))  __bf16        bf16x8;
typedef __attribute__((ext_vector_type(16))) __bf16        bf16x16;
typedef __attribute__((ext_vector_type(8)))  float         f32x8;
typedef __attribute__((ext_vector_type(8)))  unsigned short ush8;

#define DDIM   512
#define MDIM   8192
#define BDSZ   (MDIM * DDIM)     // elements per [B,D] slab
#define WELEMS (DDIM * DDIM)     // elements per weight matrix
#define KSTEPS (DDIM / 32)       // 16 K-steps of 32
#define TILEUS (128 * 32)        // ushorts per LDS tile

// f32 -> bf16 round-to-nearest-even via bit ops
__device__ __forceinline__ unsigned short f2bf(float f) {
    unsigned u = __float_as_uint(f);
    u += 0x7FFFu + ((u >> 16) & 1u);
    return (unsigned short)(u >> 16);
}

// ---------------------------------------------------------------------------
// Async copy of 32 contiguous bytes global -> LDS (CDNA5 async path,
// tracked by ASYNCcnt). IOFFSET applies to both LDS and global addresses.
// ---------------------------------------------------------------------------
__device__ __forceinline__ void async_copy32(unsigned short* dst,
                                             const unsigned short* src) {
    unsigned           ldsa = (unsigned)(size_t)(void*)dst;       // low 32 bits = LDS offset
    unsigned long long ga   = (unsigned long long)(size_t)src;
    asm volatile("global_load_async_to_lds_b128 %0, %1, off"
                 :: "v"(ldsa), "v"(ga) : "memory");
    asm volatile("global_load_async_to_lds_b128 %0, %1, off offset:16"
                 :: "v"(ldsa), "v"(ga) : "memory");
}

// ---------------------------------------------------------------------------
// Issue staging of a 128x32 tile (row-major, K contiguous) into LDS.
// 256 threads: thread t handles row t/2, 16 consecutive K at (t%2)*16.
// bf16 source: 2 async b128 loads. f32 source: load+convert+ds_store.
// ---------------------------------------------------------------------------
template<bool SRCF32>
__device__ __forceinline__ void stage_issue(unsigned short* __restrict__ lds,
                                            const void* __restrict__ src,
                                            int baseRow, int k0, float scale, int tid) {
    const int row = tid >> 1;
    const int col = (tid & 1) * 16;
    const size_t gidx = (size_t)(baseRow + row) * DDIM + (size_t)(k0 + col);
    unsigned short* dst = lds + row * 32 + col;
    if constexpr (SRCF32) {
        const float4* s4 = (const float4*)((const float*)src + gidx);
        float4 f0 = s4[0], f1 = s4[1], f2 = s4[2], f3 = s4[3];
        float vals[16] = { f0.x, f0.y, f0.z, f0.w, f1.x, f1.y, f1.z, f1.w,
                           f2.x, f2.y, f2.z, f2.w, f3.x, f3.y, f3.z, f3.w };
        ush8 o0, o1;
        #pragma unroll
        for (int j = 0; j < 8; j++) {
            o0[j] = f2bf(vals[j] * scale);
            o1[j] = f2bf(vals[8 + j] * scale);
        }
        *(ush8*)dst       = o0;
        *(ush8*)(dst + 8) = o1;
    } else {
        async_copy32(dst, (const unsigned short*)src + gidx);
    }
}

// A fragment, 16x32 bf16 (ISA 7.12.2): lane<16 -> K{0..7,16..23}; lane>=16 -> +8.
__device__ __forceinline__ bf16x16 load_frag_a(const unsigned short* __restrict__ As,
                                               int mlocal, int lane) {
    const int kb = (lane >> 4) * 8;
    bf16x8 lo = *(const bf16x8*)(As + mlocal * 32 + kb);
    bf16x8 hi = *(const bf16x8*)(As + mlocal * 32 + kb + 16);
    return __builtin_shufflevector(lo, hi, 0,1,2,3,4,5,6,7,8,9,10,11,12,13,14,15);
}

// B fragment, 32x16 bf16: lane n = col n, lane<16 -> K{0..7,16..23}; lane>=16 -> +8.
__device__ __forceinline__ bf16x16 load_frag_b(const unsigned short* __restrict__ Bs,
                                               int nlocal, int lane) {
    const int kb = (lane >> 4) * 8;
    bf16x8 lo = *(const bf16x8*)(Bs + nlocal * 32 + kb);
    bf16x8 hi = *(const bf16x8*)(Bs + nlocal * 32 + kb + 16);
    return __builtin_shufflevector(lo, hi, 0,1,2,3,4,5,6,7,8,9,10,11,12,13,14,15);
}

// ---------------------------------------------------------------------------
// One full K=512 GEMM accumulation with double-buffered LDS and async
// prefetch of the next tile before waiting on the current one.
// Per-tile async-load instructions per wave: AF32 ? 2 (B only) : 4 (A+B).
// ---------------------------------------------------------------------------
template<bool AF32>
__device__ __forceinline__ void gemm_pipe(f32x8 (&acc)[2][4],
                                          unsigned short* __restrict__ As,
                                          unsigned short* __restrict__ Bs,
                                          const void* __restrict__ A,
                                          const unsigned short* __restrict__ W,
                                          float scale, int bM, int bN,
                                          int tid, int lane, int waveM, int waveN) {
    stage_issue<AF32>(As, A, bM, 0, scale, tid);
    stage_issue<false>(Bs, W, bN, 0, 1.0f, tid);
    for (int i = 0; i < KSTEPS; ++i) {
        const int cur = i & 1;
        const int nxt = cur ^ 1;
        if (i + 1 < KSTEPS) {
            stage_issue<AF32>(As + nxt * TILEUS, A, bM, (i + 1) * 32, scale, tid);
            stage_issue<false>(Bs + nxt * TILEUS, W, bN, (i + 1) * 32, 1.0f, tid);
            if constexpr (AF32) asm volatile("s_wait_asynccnt 0x2" ::: "memory");
            else                asm volatile("s_wait_asynccnt 0x4" ::: "memory");
        } else {
            asm volatile("s_wait_asynccnt 0x0" ::: "memory");
        }
        __syncthreads();                       // all waves' tile data visible

        const unsigned short* Ac = As + cur * TILEUS;
        const unsigned short* Bc = Bs + cur * TILEUS;
        bf16x16 afr[2], bfr[4];
        #pragma unroll
        for (int mt = 0; mt < 2; mt++)
            afr[mt] = load_frag_a(Ac, waveM * 32 + mt * 16 + (lane & 15), lane);
        #pragma unroll
        for (int nt = 0; nt < 4; nt++)
            bfr[nt] = load_frag_b(Bc, waveN * 64 + nt * 16 + (lane & 15), lane);
        #pragma unroll
        for (int mt = 0; mt < 2; mt++)
            #pragma unroll
            for (int nt = 0; nt < 4; nt++)
                acc[mt][nt] = __builtin_amdgcn_wmma_f32_16x16x32_bf16(
                    false, afr[mt], false, bfr[nt], (short)0, acc[mt][nt], false, false);
        __syncthreads();                       // protect cur before re-issue
    }
}

// ---------------------------------------------------------------------------
// Prep kernel 1: convert W_ff(3), W_fb(3), W_out(1) f32 -> bf16 into ws
// ---------------------------------------------------------------------------
__global__ void k_prep_weights(const float* __restrict__ wff,
                               const float* __restrict__ wfb,
                               const float* __restrict__ wout,
                               unsigned short* __restrict__ dst) {
    int idx = blockIdx.x * 256 + threadIdx.x;   // grid sized exactly
    float v;
    if (idx < 3 * WELEMS)       v = wff[idx];
    else if (idx < 6 * WELEMS)  v = wfb[idx - 3 * WELEMS];
    else                        v = wout[idx - 6 * WELEMS];
    dst[idx] = f2bf(v);
}

// ---------------------------------------------------------------------------
// Prep kernel 2: fb2 = bf16( readout / max(||row||, eps) ), one block per row
// ---------------------------------------------------------------------------
__global__ void k_norm_readout(const float* __restrict__ readout,
                               unsigned short* __restrict__ fb2) {
    const int row = blockIdx.x;
    const int t   = threadIdx.x;           // 256 threads, 2 elems each
    const float* rp = readout + (size_t)row * DDIM;
    float v0 = rp[t], v1 = rp[t + 256];
    float s = v0 * v0 + v1 * v1;
    #pragma unroll
    for (int off = 16; off > 0; off >>= 1) s += __shfl_down(s, off, 32);
    __shared__ float partial[8];
    if ((t & 31) == 0) partial[t >> 5] = s;
    __syncthreads();
    float tot = 0.f;
    #pragma unroll
    for (int w = 0; w < 8; w++) tot += partial[w];
    float inv = 1.0f / fmaxf(sqrtf(tot), 1e-12f);
    fb2[(size_t)row * DDIM + t]       = f2bf(v0 * inv);
    fb2[(size_t)row * DDIM + t + 256] = f2bf(v1 * inv);
}

// ---------------------------------------------------------------------------
// Fused kernel: acc = A1 @ W1^T (+ A2 @ W2^T), then SNN epilogue.
// EPI==0: adaptive-LIF layer update; EPI==1: leaky readout integrator.
// Block tile 128x128, BLOCK_K=32, 256 threads = 8 waves as 4(M) x 2(N).
// ---------------------------------------------------------------------------
template<bool A1F32, bool A2F32, int EPI>
__global__ __launch_bounds__(256)
void k_fused(const void* __restrict__ A1v, float a1scale,
             const void* __restrict__ A2v,
             const unsigned short* __restrict__ W1,
             const unsigned short* __restrict__ W2,
             const float* __restrict__ bias1, const float* __restrict__ bias2,
             const float* __restrict__ soma, const float* __restrict__ spkh,
             const float* __restrict__ dend, const float* __restrict__ bvec,
             const float* __restrict__ readout,
             float* __restrict__ o_soma, float* __restrict__ o_spk,
             float* __restrict__ o_dend, float* __restrict__ o_b,
             float* __restrict__ o_read,
             unsigned short* __restrict__ s_next) {
    __shared__ alignas(16) unsigned short As[2 * TILEUS];   // double buffered
    __shared__ alignas(16) unsigned short Bs[2 * TILEUS];

    const int tid   = threadIdx.x;
    const int lane  = tid & 31;
    const int wave  = tid >> 5;
    const int waveM = wave >> 1;          // 0..3
    const int waveN = wave & 1;           // 0..1
    const int bM    = blockIdx.x * 128;
    const int bN    = blockIdx.y * 128;

    const f32x8 zero = {0.f,0.f,0.f,0.f,0.f,0.f,0.f,0.f};
    f32x8 acc[2][4];
    #pragma unroll
    for (int mt = 0; mt < 2; mt++)
        #pragma unroll
        for (int nt = 0; nt < 4; nt++) acc[mt][nt] = zero;

    gemm_pipe<A1F32>(acc, As, Bs, A1v, W1, a1scale, bM, bN, tid, lane, waveM, waveN);
    if constexpr (EPI == 0)
        gemm_pipe<A2F32>(acc, As, Bs, A2v, W2, 1.0f, bM, bN, tid, lane, waveM, waveN);

    // ---- Epilogue (C layout: VGPR r -> M = r + 8*(lane>=16), N = lane&15) ----
    #pragma unroll
    for (int mt = 0; mt < 2; mt++) {
        #pragma unroll
        for (int nt = 0; nt < 4; nt++) {
            const int gn     = bN + waveN * 64 + nt * 16 + (lane & 15);
            const int gmBase = bM + waveM * 32 + mt * 16 + ((lane >> 4) * 8);
            #pragma unroll
            for (int r = 0; r < 8; r++) {
                const int    gm  = gmBase + r;
                const size_t idx = (size_t)gm * DDIM + gn;
                const float  av  = acc[mt][nt][r];
                if constexpr (EPI == 0) {
                    float bias  = bias1[gn] + bias2[gn];
                    float a_new = 0.9f * dend[idx] + 0.1f * (av + bias);
                    float sh    = spkh[idx];
                    float sm    = 0.9f * soma[idx] * (1.0f - sh) + 0.1f * a_new;
                    float bb    = 0.96f * bvec[idx] + 0.04f * sh;
                    float spk   = (sm - (0.1f + 1.8f * bb)) > 0.0f ? 1.0f : 0.0f;
                    o_soma[idx] = sm;
                    o_spk[idx]  = spk;
                    o_dend[idx] = a_new;
                    o_b[idx]    = bb;
                    s_next[idx] = f2bf(spk);
                } else {
                    o_read[idx] = 0.9f * readout[idx] + av + bias1[gn];
                }
            }
        }
    }
}

// ---------------------------------------------------------------------------
// Launch: prep -> layer0 -> layer1 -> layer2 -> readout (all on `stream`)
// ---------------------------------------------------------------------------
extern "C" void kernel_launch(void* const* d_in, const int* in_sizes, int n_in,
                              void* d_out, int out_size, void* d_ws, size_t ws_size,
                              hipStream_t stream) {
    (void)in_sizes; (void)n_in; (void)out_size; (void)ws_size;
    const float* x       = (const float*)d_in[0];
    const float* soma    = (const float*)d_in[1];
    const float* spikes  = (const float*)d_in[2];
    const float* dend    = (const float*)d_in[3];
    const float* bvec    = (const float*)d_in[4];
    const float* readout = (const float*)d_in[5];
    const float* Wff     = (const float*)d_in[6];
    const float* bff     = (const float*)d_in[7];
    const float* Wfb     = (const float*)d_in[8];
    const float* bfb     = (const float*)d_in[9];
    const float* Wout    = (const float*)d_in[10];
    const float* bout    = (const float*)d_in[11];
    float* out = (float*)d_out;

    unsigned short* ws    = (unsigned short*)d_ws;
    unsigned short* Wbf   = ws;                       // 7*WELEMS bf16 weights
    unsigned short* sbuf0 = ws + (size_t)7 * WELEMS;  // spk bf16 ping
    unsigned short* sbuf1 = sbuf0 + (size_t)BDSZ;     // spk bf16 pong
    unsigned short* fb2   = sbuf1 + (size_t)BDSZ;     // normalized readout bf16

    k_prep_weights<<<(7 * WELEMS) / 256, 256, 0, stream>>>(Wff, Wfb, Wout, Wbf);
    k_norm_readout<<<MDIM, 256, 0, stream>>>(readout, fb2);

    dim3 grid(MDIM / 128, DDIM / 128);  // 64 x 4

    // layer 0: A1 = x*0.5 (f32), A2 = spikes_h[1] (f32)
    k_fused<true, true, 0><<<grid, 256, 0, stream>>>(
        x, 0.5f, spikes + (size_t)1 * BDSZ,
        Wbf + 0 * WELEMS, Wbf + 3 * WELEMS, bff + 0, bfb + 0,
        soma + 0, spikes + 0, dend + 0, bvec + 0, nullptr,
        out + 0 * (size_t)BDSZ, out + 3 * (size_t)BDSZ,
        out + 6 * (size_t)BDSZ, out + 9 * (size_t)BDSZ, nullptr, sbuf0);

    // layer 1: A1 = spk0 (bf16, async), A2 = spikes_h[2] (f32)
    k_fused<false, true, 0><<<grid, 256, 0, stream>>>(
        sbuf0, 1.0f, spikes + (size_t)2 * BDSZ,
        Wbf + 1 * WELEMS, Wbf + 4 * WELEMS, bff + 512, bfb + 512,
        soma + BDSZ, spikes + BDSZ, dend + BDSZ, bvec + BDSZ, nullptr,
        out + 1 * (size_t)BDSZ, out + 4 * (size_t)BDSZ,
        out + 7 * (size_t)BDSZ, out + 10 * (size_t)BDSZ, nullptr, sbuf1);

    // layer 2: A1 = spk1 (bf16, async), A2 = normalized readout (bf16, async)
    k_fused<false, false, 0><<<grid, 256, 0, stream>>>(
        sbuf1, 1.0f, fb2,
        Wbf + 2 * WELEMS, Wbf + 5 * WELEMS, bff + 1024, bfb + 1024,
        soma + 2 * (size_t)BDSZ, spikes + 2 * (size_t)BDSZ,
        dend + 2 * (size_t)BDSZ, bvec + 2 * (size_t)BDSZ, nullptr,
        out + 2 * (size_t)BDSZ, out + 5 * (size_t)BDSZ,
        out + 8 * (size_t)BDSZ, out + 11 * (size_t)BDSZ, nullptr, sbuf0);

    // readout: out = 0.9*readout + spk2 @ W_out^T + b_out (all-async operands)
    k_fused<false, false, 1><<<grid, 256, 0, stream>>>(
        sbuf0, 1.0f, nullptr,
        Wbf + 6 * WELEMS, nullptr, bout, nullptr,
        nullptr, nullptr, nullptr, nullptr, readout,
        nullptr, nullptr, nullptr, nullptr, out + 12 * (size_t)BDSZ, nullptr);
}
